// SiteUpdate_1855425871939
// MI455X (gfx1250) — matrix operand order repaired and checked by reading, not compiled, not run on hardware
//
#include <hip/hip_runtime.h>
#include <hip/hip_bf16.h>

typedef __attribute__((ext_vector_type(16))) __bf16 v16bf;
typedef __attribute__((ext_vector_type(8)))  float  v8f;
typedef unsigned short ushort_t;

#define NB   16        // batch
#define NS   2048      // sites
#define NBOND 65536
#define FEAT 64
#define DIN  192
#define H1D  256
#define H2D  256
#define DOUT 64
#define MROWS (NB*NS)  // 32768

#define BM 256
#define BN 64
#define KT 32
#define LDAP (KT + 8)  // padded LDS stride in halves (80B, 16B aligned)

__device__ __forceinline__ ushort_t f2bf(float f) {
    union { float f; unsigned u; } x; x.f = f;
    unsigned r = x.u + 0x7FFFu + ((x.u >> 16) & 1u);   // round-to-nearest-even
    return (ushort_t)(r >> 16);
}

// CDNA5 async global->LDS copy (ASYNCcnt-tracked), ISA 15.18.3 op 98.
__device__ __forceinline__ void async_ld_b128(unsigned lds_off, const void* g) {
    asm volatile("global_load_async_to_lds_b128 %0, %1, off"
                 :: "v"(lds_off), "v"(g) : "memory");
}
__device__ __forceinline__ void wait_async0() {
    asm volatile("s_wait_asynccnt 0x0" ::: "memory");
}
__device__ __forceinline__ unsigned lds_off32(const void* p) {
    return (unsigned)(unsigned long long)p;   // generic LDS addr: [31:0] = offset
}

// ---------------------------------------------------------------------------
// 1) scatter-add: pool[b][site][f] += bonds[b][j][f]; counts[site] += 1
// ---------------------------------------------------------------------------
__global__ __launch_bounds__(256)
void scatter_add_kernel(const float* __restrict__ bonds,
                        const int*   __restrict__ idx,
                        float* __restrict__ pool,
                        float* __restrict__ counts) {
    int t = blockIdx.x * blockDim.x + threadIdx.x;   // 65536*64 threads
    int j = t >> 6;
    int f = t & 63;
    int s = idx[j];
    #pragma unroll 4
    for (int b = 0; b < NB; ++b) {
        float v = bonds[((size_t)b * NBOND + j) * FEAT + f];
        atomicAdd(&pool[((size_t)b * NS + s) * FEAT + f], v);
    }
    if (f == 0) atomicAdd(&counts[s], 1.0f);
}

// ---------------------------------------------------------------------------
// 2) finalize mean + concat [pool | sites | states] -> bf16 V[32768][192]
// ---------------------------------------------------------------------------
__global__ __launch_bounds__(256)
void concat_kernel(const float* __restrict__ pool,
                   const float* __restrict__ counts,
                   const float* __restrict__ sites,
                   const float* __restrict__ states,
                   ushort_t* __restrict__ V) {
    int t = blockIdx.x * blockDim.x + threadIdx.x;   // 32768*192 threads
    int row = t / DIN;
    int c   = t - row * DIN;
    int b = row >> 11;
    int s = row & 2047;
    float v;
    if (c < FEAT) {
        float cnt = counts[s];
        float den = cnt > 1.0f ? cnt : 1.0f;
        v = pool[((size_t)b * NS + s) * FEAT + c] / den;
    } else if (c < 2 * FEAT) {
        v = sites[((size_t)b * NS + s) * FEAT + (c - FEAT)];
    } else {
        v = states[(size_t)b * FEAT + (c - 2 * FEAT)];
    }
    V[(size_t)row * DIN + c] = f2bf(v);
}

// ---------------------------------------------------------------------------
// 3) weight prep: W[K][N] f32  ->  Wt[N][K] bf16
// ---------------------------------------------------------------------------
__global__ __launch_bounds__(256)
void transpose_w_kernel(const float* __restrict__ W, ushort_t* __restrict__ Wt,
                        int K, int N) {
    int t = blockIdx.x * blockDim.x + threadIdx.x;
    if (t >= K * N) return;
    int k = t / N;
    int n = t - k * N;
    Wt[(size_t)n * K + k] = f2bf(W[(size_t)k * N + n]);
}

// ---------------------------------------------------------------------------
// 4) WMMA bf16 GEMM, double-buffered async staging, fully unrolled K loop.
//    C[M][N] = relu(A[M][K] @ Wt^T + bias)
//    block = 256 threads (8 waves), tile BM x BN = 256 x 64,
//    wave w -> rows [w*32, w*32+32) x all 64 cols = 2x4 WMMA fragments.
// ---------------------------------------------------------------------------
template <int M, int N, int K, bool OUT_BF16>
__global__ __launch_bounds__(256)
void gemm_bias_relu_kernel(const ushort_t* __restrict__ A,
                           const ushort_t* __restrict__ Wt,
                           const float* __restrict__ bias,
                           void* __restrict__ Cout) {
    __shared__ __align__(16) ushort_t sA[2][BM * LDAP];   // 2 x 20 KB
    __shared__ __align__(16) ushort_t sB[2][BN * LDAP];   // 2 x 5 KB

    const int tid  = threadIdx.x;
    const int lane = tid & 31;
    const int wave = tid >> 5;
    const int bm = blockIdx.x * BM;
    const int bn = blockIdx.y * BN;
    const int wm = wave << 5;         // 0..224

    const int lgrp = lane >> 4;       // K-half select
    const int lidx = lane & 15;       // row (A) / col (B) within fragment
    const int kbA  = lgrp * 8;        // A: lanes16-31 hold K=8..15 / 24..31
    const int kbB  = lgrp * 8;        // B: lanes16-31 hold K=8..15 / 24..31

    // per-thread staging coordinates (4x uint4 for A, 1x uint4 for B)
    const int arow[4] = { (tid + 0) >> 2, (tid + 256) >> 2,
                          (tid + 512) >> 2, (tid + 768) >> 2 };
    const int aseg = tid & 3;
    const int brow = tid >> 2;
    const int bseg = tid & 3;

    v8f acc[2][4] = {};

    constexpr int nk = K / KT;

    // stage tile 0 into buffer 0 (async copy engine, no VGPR round-trip)
    #pragma unroll
    for (int i = 0; i < 4; ++i)
        async_ld_b128(lds_off32(&sA[0][arow[i] * LDAP + aseg * 8]),
                      A + (size_t)(bm + arow[i]) * K + aseg * 8);
    async_ld_b128(lds_off32(&sB[0][brow * LDAP + bseg * 8]),
                  Wt + (size_t)(bn + brow) * K + bseg * 8);
    wait_async0();
    __syncthreads();

    #pragma unroll
    for (int kt = 0; kt < nk; ++kt) {
        const int p = kt & 1;
        // prefetch next K-tile into the other buffer while we compute
        if (kt + 1 < nk) {
            const int kk = (kt + 1) * KT;
            #pragma unroll
            for (int i = 0; i < 4; ++i)
                async_ld_b128(lds_off32(&sA[1 - p][arow[i] * LDAP + aseg * 8]),
                              A + (size_t)(bm + arow[i]) * K + kk + aseg * 8);
            async_ld_b128(lds_off32(&sB[1 - p][brow * LDAP + bseg * 8]),
                          Wt + (size_t)(bn + brow) * K + kk + bseg * 8);
        }

        // load ALL fragments into distinct registers first so the compiler
        // can clause the ds_loads and use partial s_wait_dscnt, overlapping
        // WMMA execution with remaining LDS reads.
        union Frag { v16bf v; uint4 u[2]; };
        Frag fa[2], fb[4];
        #pragma unroll
        for (int fm = 0; fm < 2; ++fm) {
            const ushort_t* ap = &sA[p][(wm + fm * 16 + lidx) * LDAP];
            fa[fm].u[0] = *(const uint4*)(ap + kbA);
            fa[fm].u[1] = *(const uint4*)(ap + 16 + kbA);
        }
        #pragma unroll
        for (int fn = 0; fn < 4; ++fn) {
            const ushort_t* bp = &sB[p][(fn * 16 + lidx) * LDAP + kbB];
            fb[fn].u[0] = *(const uint4*)(bp);
            fb[fn].u[1] = *(const uint4*)(bp + 16);
        }
        #pragma unroll
        for (int fn = 0; fn < 4; ++fn) {
            acc[0][fn] = __builtin_amdgcn_wmma_f32_16x16x32_bf16(
                false, fa[0].v, false, fb[fn].v, (short)0, acc[0][fn], false, false);
            acc[1][fn] = __builtin_amdgcn_wmma_f32_16x16x32_bf16(
                false, fa[1].v, false, fb[fn].v, (short)0, acc[1][fn], false, false);
        }

        if (kt + 1 < nk) {
            wait_async0();      // our async copies into buf 1-p are done
            __syncthreads();    // everyone's copies visible, reads of buf p done
        }
    }

    // epilogue: C/D layout -> VGPR i: M = base + 8*lgrp + i, N = base + lidx
    #pragma unroll
    for (int fm = 0; fm < 2; ++fm) {
        #pragma unroll
        for (int fn = 0; fn < 4; ++fn) {
            int col = bn + fn * 16 + lidx;
            float bv = bias[col];
            #pragma unroll
            for (int i = 0; i < 8; ++i) {
                int row = bm + wm + fm * 16 + lgrp * 8 + i;
                float v = acc[fm][fn][i] + bv;
                v = v > 0.0f ? v : 0.0f;
                if (OUT_BF16)
                    ((ushort_t*)Cout)[(size_t)row * N + col] = f2bf(v);
                else
                    ((float*)Cout)[(size_t)row * N + col] = v;
            }
        }
    }
}

// ---------------------------------------------------------------------------
extern "C" void kernel_launch(void* const* d_in, const int* in_sizes, int n_in,
                              void* d_out, int out_size, void* d_ws, size_t ws_size,
                              hipStream_t stream) {
    const float* sites  = (const float*)d_in[0];
    const float* bonds  = (const float*)d_in[1];
    const float* states = (const float*)d_in[2];
    const int*   idx    = (const int*)  d_in[3];
    const float* W1 = (const float*)d_in[4];
    const float* b1 = (const float*)d_in[5];
    const float* W2 = (const float*)d_in[6];
    const float* b2 = (const float*)d_in[7];
    const float* W3 = (const float*)d_in[8];
    const float* b3 = (const float*)d_in[9];

    char* ws = (char*)d_ws;
    size_t off = 0;
    auto carve = [&](size_t bytes) -> char* {
        char* p = ws + off;
        off = (off + bytes + 255) & ~(size_t)255;
        return p;
    };
    float*    pool   = (float*)   carve((size_t)NB * NS * FEAT * 4);
    float*    counts = (float*)   carve((size_t)NS * 4);
    ushort_t* V      = (ushort_t*)carve((size_t)MROWS * DIN * 2);
    ushort_t* H1buf  = (ushort_t*)carve((size_t)MROWS * H1D * 2);
    ushort_t* H2buf  = (ushort_t*)carve((size_t)MROWS * H2D * 2);
    ushort_t* Wt1    = (ushort_t*)carve((size_t)H1D * DIN * 2);
    ushort_t* Wt2    = (ushort_t*)carve((size_t)H2D * H1D * 2);
    ushort_t* Wt3    = (ushort_t*)carve((size_t)DOUT * H2D * 2);
    (void)ws_size;

    hipMemsetAsync(pool,   0, (size_t)NB * NS * FEAT * 4, stream);
    hipMemsetAsync(counts, 0, (size_t)NS * 4, stream);

    scatter_add_kernel<<<(NBOND * FEAT) / 256, 256, 0, stream>>>(bonds, idx, pool, counts);
    concat_kernel<<<((size_t)MROWS * DIN) / 256, 256, 0, stream>>>(pool, counts, sites, states, V);

    transpose_w_kernel<<<(DIN * H1D + 255) / 256, 256, 0, stream>>>(W1, Wt1, DIN, H1D);
    transpose_w_kernel<<<(H1D * H2D + 255) / 256, 256, 0, stream>>>(W2, Wt2, H1D, H2D);
    transpose_w_kernel<<<(H2D * DOUT + 255) / 256, 256, 0, stream>>>(W3, Wt3, H2D, DOUT);

    dim3 blk(256);
    dim3 g1(MROWS / BM, H1D / BN);
    gemm_bias_relu_kernel<MROWS, H1D, DIN, true ><<<g1, blk, 0, stream>>>(V,     Wt1, b1, H1buf);
    dim3 g2(MROWS / BM, H2D / BN);
    gemm_bias_relu_kernel<MROWS, H2D, H1D, true ><<<g2, blk, 0, stream>>>(H1buf, Wt2, b2, H2buf);
    dim3 g3(MROWS / BM, DOUT / BN);
    gemm_bias_relu_kernel<MROWS, DOUT, H2D, false><<<g3, blk, 0, stream>>>(H2buf, Wt3, b3, d_out);
}
